// Attention_76639396430017
// MI455X (gfx1250) — matrix operand order repaired and checked by reading, not compiled, not run on hardware
//
#include <hip/hip_runtime.h>
#include <hip/hip_bf16.h>

typedef __attribute__((ext_vector_type(16))) _Float16 v16h;
typedef __attribute__((ext_vector_type(8)))  _Float16 v8h;
typedef __attribute__((ext_vector_type(8)))  float    v8f;

#define B_    2
#define S_    2048
#define H_    4096
#define NH_   32
#define NKV_  8
#define HD_   128
#define MTOT  (B_ * S_)                // 4096 rows of x
#define NQKV  ((NH_ + 2 * NKV_) * HD_) // 6144 fused qkv output cols

// ---------------- async global->LDS staging (CDNA5 ASYNCcnt path) ----------

#if defined(__has_builtin)
#if __has_builtin(__builtin_amdgcn_global_load_async_to_lds_b128)
#define HAVE_ASYNC_LDS 1
#endif
#endif

#ifdef HAVE_ASYNC_LDS
// builtin signature (from clang diagnostic): param0 is
// 'int __attribute__((vector_size(16))) __device__ *'  i.e. AS(1) int4*
typedef int v4i_t __attribute__((vector_size(16)));
typedef __attribute__((address_space(1))) v4i_t gv4i;
typedef __attribute__((address_space(3))) v4i_t lv4i;
#endif

// copy 16 f16 (32 bytes) from global to LDS
__device__ __forceinline__ void stage16(const _Float16* g, _Float16* l) {
#ifdef HAVE_ASYNC_LDS
  __builtin_amdgcn_global_load_async_to_lds_b128((gv4i*)g, (lv4i*)l, 0, 0);
  __builtin_amdgcn_global_load_async_to_lds_b128((gv4i*)(g + 8),
                                                 (lv4i*)(l + 8), 0, 0);
#else
  *(v8h*)l = *(const v8h*)g;
  *(v8h*)(l + 8) = *(const v8h*)(g + 8);
#endif
}

__device__ __forceinline__ void wait_async0() {
#ifdef HAVE_ASYNC_LDS
#if __has_builtin(__builtin_amdgcn_s_wait_asynccnt)
  __builtin_amdgcn_s_wait_asynccnt(0);
#else
  asm volatile("s_wait_asynccnt 0x0" ::: "memory");
#endif
#endif
}

// ---------------- WMMA fragment helpers (CDNA5 wave32 layouts) --------------

// A-matrix 16x32 f16 (MxK): lane L<16 -> row M=L, K=0..15;
// lane L>=16 -> row M=L-16, K=16..31. Two contiguous v8h loads.
__device__ __forceinline__ v16h load_a_frag(const _Float16* base, int ld) {
  const int lane = threadIdx.x & 31;
  const int g = lane >> 4, row = lane & 15;
  const _Float16* p = base + row * ld + 16 * g;
  v8h lo = *(const v8h*)(p);
  v8h hi = *(const v8h*)(p + 8);
  return __builtin_shufflevector(lo, hi, 0, 1, 2, 3, 4, 5, 6, 7,
                                 8, 9, 10, 11, 12, 13, 14, 15);
}

// B-matrix 32x16 f16 (KxN) sourced from storage laid out [N][K]:
// lanes 0-15 hold K=0..15 for col N=lane, lanes 16-31 hold K=16..31.
__device__ __forceinline__ v16h load_b_frag_nk(const _Float16* base, int ld) {
  const int lane = threadIdx.x & 31;
  const int g = lane >> 4, n = lane & 15;
  const _Float16* p = base + n * ld + 16 * g;
  v8h lo = *(const v8h*)(p);
  v8h hi = *(const v8h*)(p + 8);
  return __builtin_shufflevector(lo, hi, 0, 1, 2, 3, 4, 5, 6, 7,
                                 8, 9, 10, 11, 12, 13, 14, 15);
}

// B-matrix 32x16 f16 (KxN) sourced from storage laid out [K][N] (V tile).
__device__ __forceinline__ v16h load_b_frag_kn(const _Float16* base, int ld) {
  const int lane = threadIdx.x & 31;
  const int g = lane >> 4, n = lane & 15;
  const _Float16* p = base + (16 * g) * ld + n;
  v16h b;
#pragma unroll
  for (int v = 0; v < 8; ++v) {
    b[2 * v]     = p[(2 * v) * ld];
    b[2 * v + 1] = p[(2 * v + 1) * ld];
  }
  return b;
}

__device__ __forceinline__ v8f wmma_f16(v16h a, v16h b, v8f c) {
  return __builtin_amdgcn_wmma_f32_16x16x32_f16(false, a, false, b,
                                                (short)0, c, false, false);
}

// ---------------- fp32 -> f16 conversion ----------------

__global__ void cvt_f32_f16(const float* __restrict__ in,
                            _Float16* __restrict__ out, int n) {
  int i = blockIdx.x * blockDim.x + threadIdx.x;
  const int stride = gridDim.x * blockDim.x;
  for (; i < n; i += stride) out[i] = (_Float16)in[i];
}

// ---------------- GEMM: C[M,N] = A[M,K] * Bw[N,K]^T (f16 in, f32 out) -------
// Block tile 128x128, K-step 32, 256 threads = 8 waves (4 along M, 2 along N),
// wave tile 32x64 = 2x4 WMMA accumulators. LDS double-buffered; next tile is
// streamed (async global->LDS when available) while WMMAs run on the current.

__global__ __launch_bounds__(256) void gemm_f16f32(
    const _Float16* __restrict__ A, const _Float16* __restrict__ Bw,
    float* __restrict__ C, int M, int N, int K) {
  __shared__ __align__(32) _Float16 As[2][128][32];  // 16 KB
  __shared__ __align__(32) _Float16 Bs[2][128][32];  // 16 KB
  const int tid = threadIdx.x;
  const int bm = blockIdx.y * 128, bn = blockIdx.x * 128;
  const int wave = tid >> 5, lane = tid & 31;
  const int wm = (wave & 3) * 32, wn = (wave >> 2) * 64;
  const int row = tid >> 1, col = (tid & 1) * 16;
  const _Float16* aptr = A + (size_t)(bm + row) * K + col;
  const _Float16* bptr = Bw + (size_t)(bn + row) * K + col;

  v8f acc[2][4] = {};
  const int nkt = K / 32;

  stage16(aptr, &As[0][row][col]);
  stage16(bptr, &Bs[0][row][col]);

  for (int kt = 0; kt < nkt; ++kt) {
    wait_async0();
    __syncthreads();
    if (kt + 1 < nkt) {  // stream next K-tile into the other buffer
      stage16(aptr + (kt + 1) * 32, &As[(kt + 1) & 1][row][col]);
      stage16(bptr + (kt + 1) * 32, &Bs[(kt + 1) & 1][row][col]);
    }
    const int bb = kt & 1;
    v16h a0 = load_a_frag(&As[bb][wm][0], 32);
    v16h a1 = load_a_frag(&As[bb][wm + 16][0], 32);
    v16h b0 = load_b_frag_nk(&Bs[bb][wn][0], 32);
    v16h b1 = load_b_frag_nk(&Bs[bb][wn + 16][0], 32);
    v16h b2 = load_b_frag_nk(&Bs[bb][wn + 32][0], 32);
    v16h b3 = load_b_frag_nk(&Bs[bb][wn + 48][0], 32);
    acc[0][0] = wmma_f16(a0, b0, acc[0][0]);
    acc[0][1] = wmma_f16(a0, b1, acc[0][1]);
    acc[0][2] = wmma_f16(a0, b2, acc[0][2]);
    acc[0][3] = wmma_f16(a0, b3, acc[0][3]);
    acc[1][0] = wmma_f16(a1, b0, acc[1][0]);
    acc[1][1] = wmma_f16(a1, b1, acc[1][1]);
    acc[1][2] = wmma_f16(a1, b2, acc[1][2]);
    acc[1][3] = wmma_f16(a1, b3, acc[1][3]);
  }

  const int g = lane >> 4, n = lane & 15;
#pragma unroll
  for (int i = 0; i < 2; ++i)
#pragma unroll
    for (int j = 0; j < 4; ++j)
#pragma unroll
      for (int r = 0; r < 8; ++r)
        C[(size_t)(bm + wm + 16 * i + 8 * g + r) * N +
          (bn + wn + 16 * j + n)] = acc[i][j][r];
}

// ---------------- rotary + head split ----------------

__global__ void rotary_split(const float* __restrict__ qkv,
                             const float* __restrict__ fc,
                             const float* __restrict__ fs,
                             _Float16* __restrict__ qh,
                             _Float16* __restrict__ kh,
                             _Float16* __restrict__ vh) {
  const int PAIRS = NQKV / 2;  // 3072
  int t = blockIdx.x * blockDim.x + threadIdx.x;
  if (t >= MTOT * PAIRS) return;
  const int m = t / PAIRS, nn = (t % PAIRS) * 2;
  const int b = m / S_, spos = m % S_;
  const float v0 = qkv[(size_t)m * NQKV + nn];
  const float v1 = qkv[(size_t)m * NQKV + nn + 1];
  if (nn < NH_ * HD_) {  // Q + rotary
    const int head = nn / HD_, d = nn % HD_;
    const float c = fc[((size_t)b * S_ + spos) * (HD_ / 2) + d / 2];
    const float s = fs[((size_t)b * S_ + spos) * (HD_ / 2) + d / 2];
    const size_t o = (((size_t)b * NH_ + head) * S_ + spos) * HD_ + d;
    qh[o]     = (_Float16)(v0 * c - v1 * s);
    qh[o + 1] = (_Float16)(v0 * s + v1 * c);
  } else if (nn < (NH_ + NKV_) * HD_) {  // K + rotary
    const int nk = nn - NH_ * HD_;
    const int head = nk / HD_, d = nk % HD_;
    const float c = fc[((size_t)b * S_ + spos) * (HD_ / 2) + d / 2];
    const float s = fs[((size_t)b * S_ + spos) * (HD_ / 2) + d / 2];
    const size_t o = (((size_t)b * NKV_ + head) * S_ + spos) * HD_ + d;
    kh[o]     = (_Float16)(v0 * c - v1 * s);
    kh[o + 1] = (_Float16)(v0 * s + v1 * c);
  } else {  // V passthrough
    const int nv = nn - (NH_ + NKV_) * HD_;
    const int head = nv / HD_, d = nv % HD_;
    const size_t o = (((size_t)b * NKV_ + head) * S_ + spos) * HD_ + d;
    vh[o]     = (_Float16)v0;
    vh[o + 1] = (_Float16)v1;
  }
}

// ---------------- flash attention ----------------
// Grid (S/64, NH, B), 128 threads = 4 waves; each wave owns 16 q rows.
// K/V tiles double-buffered; next tile streams while WMMAs run.

__global__ __launch_bounds__(128) void flash_attn(
    const _Float16* __restrict__ qh, const _Float16* __restrict__ kh,
    const _Float16* __restrict__ vh, _Float16* __restrict__ oh) {
  __shared__ __align__(32) _Float16 Qs[64][HD_];        // 16 KB
  __shared__ __align__(32) _Float16 Ks[2][32][HD_];     // 16 KB
  __shared__ __align__(32) _Float16 Vs[2][32][HD_];     // 16 KB
  __shared__ __align__(32) _Float16 Ps[4][16][32];      //  4 KB
  const int tid = threadIdx.x;
  const int w = tid >> 5, lane = tid & 31;
  const int g = lane >> 4, n = lane & 15;
  const int q0 = blockIdx.x * 64;
  const int h = blockIdx.y, b = blockIdx.z;
  const int kvh = h / (NH_ / NKV_);
  const _Float16* qbase = qh + (((size_t)b * NH_ + h) * S_ + q0) * HD_;
  const _Float16* kbase = kh + ((size_t)b * NKV_ + kvh) * S_ * HD_;
  const _Float16* vbase = vh + ((size_t)b * NKV_ + kvh) * S_ * HD_;

  // stage Q tile (8192 halfs, flat) + first K/V tile (4096 halfs each)
#pragma unroll
  for (int i = 0; i < 4; ++i)
    stage16(qbase + tid * 64 + i * 16, (_Float16*)Qs + tid * 64 + i * 16);
  stage16(kbase + tid * 32, (_Float16*)Ks[0] + tid * 32);
  stage16(kbase + tid * 32 + 16, (_Float16*)Ks[0] + tid * 32 + 16);
  stage16(vbase + tid * 32, (_Float16*)Vs[0] + tid * 32);
  stage16(vbase + tid * 32 + 16, (_Float16*)Vs[0] + tid * 32 + 16);

  float m_r[8], l_r[8], alpha[8];
  v8f Oacc[8] = {};
#pragma unroll
  for (int r = 0; r < 8; ++r) { m_r[r] = -1e30f; l_r[r] = 0.0f; }
  const float scale = 0.08838834764831845f;  // 1/sqrt(128)

  const int nkt = q0 / 32 + 2;  // causal: k tiles up to q0+63
  for (int kt = 0; kt < nkt; ++kt) {
    const int k0 = kt * 32;
    wait_async0();
    __syncthreads();
    if (kt + 1 < nkt) {  // stream next K/V tile into the other buffer
      const size_t goff = (size_t)(k0 + 32) * HD_ + tid * 32;
      _Float16* kl = (_Float16*)Ks[(kt + 1) & 1] + tid * 32;
      _Float16* vl = (_Float16*)Vs[(kt + 1) & 1] + tid * 32;
      stage16(kbase + goff, kl);
      stage16(kbase + goff + 16, kl + 16);
      stage16(vbase + goff, vl);
      stage16(vbase + goff + 16, vl + 16);
    }
    const int bb = kt & 1;

    // scores S = Q(16x128) * K^T -> two 16x16 tiles, contraction over d
    v8f s0acc = {}, s1acc = {};
#pragma unroll
    for (int dk = 0; dk < 4; ++dk) {
      v16h aq  = load_a_frag(&Qs[w * 16][dk * 32], HD_);
      v16h bk0 = load_b_frag_nk(&Ks[bb][0][dk * 32], HD_);
      v16h bk1 = load_b_frag_nk(&Ks[bb][16][dk * 32], HD_);
      s0acc = wmma_f16(aq, bk0, s0acc);
      s1acc = wmma_f16(aq, bk1, s1acc);
    }

    // online softmax; each C-layout row (M = 8g+r) lives across one 16-lane half
#pragma unroll
    for (int r = 0; r < 8; ++r) {
      const int qrow = q0 + w * 16 + 8 * g + r;
      float s0 = (k0 + n      <= qrow) ? s0acc[r] * scale : -1e9f;
      float s1 = (k0 + 16 + n <= qrow) ? s1acc[r] * scale : -1e9f;
      float pm = fmaxf(s0, s1);
#pragma unroll
      for (int off = 1; off < 16; off <<= 1)
        pm = fmaxf(pm, __shfl_xor(pm, off, 32));
      const float mn = fmaxf(m_r[r], pm);
      const float p0 = __expf(s0 - mn), p1 = __expf(s1 - mn);
      float ps = p0 + p1;
#pragma unroll
      for (int off = 1; off < 16; off <<= 1) ps += __shfl_xor(ps, off, 32);
      alpha[r] = __expf(m_r[r] - mn);
      l_r[r] = l_r[r] * alpha[r] + ps;
      m_r[r] = mn;
      Ps[w][8 * g + r][n]      = (_Float16)p0;
      Ps[w][8 * g + r][16 + n] = (_Float16)p1;
    }
    __asm__ volatile("" ::: "memory");  // order P stores before A-frag reads

    // O = O*alpha + P(16x32) * V(32x128)
    v16h ap = load_a_frag(&Ps[w][0][0], 32);
#pragma unroll
    for (int dt = 0; dt < 8; ++dt) {
      v8f o = Oacc[dt];
#pragma unroll
      for (int r = 0; r < 8; ++r) o[r] *= alpha[r];
      v16h bv = load_b_frag_kn(&Vs[bb][0][dt * 16], HD_);
      Oacc[dt] = wmma_f16(ap, bv, o);
    }
  }

  float inv_l[8];
#pragma unroll
  for (int r = 0; r < 8; ++r) inv_l[r] = 1.0f / l_r[r];
#pragma unroll
  for (int dt = 0; dt < 8; ++dt)
#pragma unroll
    for (int r = 0; r < 8; ++r) {
      const int qrow = q0 + w * 16 + 8 * g + r;
      oh[((size_t)b * S_ + qrow) * (NH_ * HD_) + h * HD_ + dt * 16 + n] =
          (_Float16)(Oacc[dt][r] * inv_l[r]);
    }
}

// ---------------- host launch ----------------

extern "C" void kernel_launch(void* const* d_in, const int* in_sizes, int n_in,
                              void* d_out, int out_size, void* d_ws,
                              size_t ws_size, hipStream_t stream) {
  const float* x  = (const float*)d_in[0];
  const float* fc = (const float*)d_in[1];
  const float* fs = (const float*)d_in[2];
  const float* wq = (const float*)d_in[3];
  const float* wk = (const float*)d_in[4];
  const float* wv = (const float*)d_in[5];
  const float* wo = (const float*)d_in[6];
  float* out = (float*)d_out;
  char* ws = (char*)d_ws;

  // workspace layout (bytes); attnh reuses the xh region (xh dead after GEMM1)
  _Float16* xh    = (_Float16*)(ws + 0);          //  32 MB  x in f16
  _Float16* attnh = (_Float16*)(ws + 0);          //  32 MB  attention out f16
  _Float16* wqkvh = (_Float16*)(ws + 33554432);   //  48 MB  [6144][4096] f16
  _Float16* woh   = (_Float16*)(ws + 83886080);   //  32 MB  [4096][4096] f16
  float*    qkv32 = (float*)(ws + 117440512);     //  96 MB  [4096][6144] f32
  _Float16* qhh   = (_Float16*)(ws + 218103808);  //  32 MB  [B,NH,S,HD] f16
  _Float16* khh   = (_Float16*)(ws + 251658240);  //   8 MB  [B,NKV,S,HD] f16
  _Float16* vhh   = (_Float16*)(ws + 260046848);  //   8 MB  [B,NKV,S,HD] f16
  // total: 268435456 bytes

  cvt_f32_f16<<<4096, 256, 0, stream>>>(x, xh, MTOT * H_);
  cvt_f32_f16<<<4096, 256, 0, stream>>>(wq, wqkvh, NH_ * HD_ * H_);
  cvt_f32_f16<<<2048, 256, 0, stream>>>(wk, wqkvh + (size_t)NH_ * HD_ * H_,
                                        NKV_ * HD_ * H_);
  cvt_f32_f16<<<2048, 256, 0, stream>>>(
      wv, wqkvh + (size_t)(NH_ + NKV_) * HD_ * H_, NKV_ * HD_ * H_);
  cvt_f32_f16<<<4096, 256, 0, stream>>>(wo, woh, H_ * NH_ * HD_);

  // fused QKV projection: [4096,6144] = xh[4096,4096] * wqkvh[6144,4096]^T
  gemm_f16f32<<<dim3(NQKV / 128, MTOT / 128), 256, 0, stream>>>(
      xh, wqkvh, qkv32, MTOT, NQKV, H_);

  const int pairs = MTOT * (NQKV / 2);
  rotary_split<<<(pairs + 255) / 256, 256, 0, stream>>>(qkv32, fc, fs, qhh,
                                                        khh, vhh);

  flash_attn<<<dim3(S_ / 64, NH_, B_), 128, 0, stream>>>(qhh, khh, vhh, attnh);

  // output projection: out[4096,4096] = attnh[4096,4096] * woh[4096,4096]^T
  gemm_f16f32<<<dim3(H_ / 128, MTOT / 128), 256, 0, stream>>>(
      attnh, woh, out, MTOT, H_, H_);

  (void)in_sizes; (void)n_in; (void)out_size; (void)ws_size;
}
